// MoELayer_40235253629369
// MI455X (gfx1250) — matrix operand order repaired and checked by reading, not compiled, not run on hardware
//
#include <hip/hip_runtime.h>
#include <math.h>

#define T_TOK 8192
#define H_DIM 1024
#define I_DIM 4096
#define N_EXP 8
#define TOPK  2
#define CAP   2560

typedef __attribute__((ext_vector_type(16))) __bf16 v16bf;
typedef __attribute__((ext_vector_type(8)))  float  v8f;

union Frag128 { uint4 q[2]; v16bf v; };

__device__ __forceinline__ unsigned short f2bf(float f) {
    unsigned int u = __float_as_uint(f);
    unsigned int r = u + 0x7FFFu + ((u >> 16) & 1u);   // round-to-nearest-even
    return (unsigned short)(r >> 16);
}

// Branch-free GELU (tanh form); indistinguishable from exact erf-GELU after
// the bf16 quantization this pipeline already applies to h.
__device__ __forceinline__ float gelu_f(float x) {
    float u = x * (0.7978845608f + 0.0356774081f * x * x);
    float t = 1.0f - 2.0f / (1.0f + __expf(2.0f * u));   // tanh(u)
    return 0.5f * x * (1.0f + t);
}

// CDNA5 async global->LDS copy (ASYNCcnt-tracked); dest is per-lane LDS byte address.
__device__ __forceinline__ void async_copy_b128(unsigned lds_addr, const void* gptr) {
    asm volatile("global_load_async_to_lds_b128 %0, %1, off"
                 :: "v"(lds_addr), "v"(gptr) : "memory");
}
__device__ __forceinline__ void wait_async0() {
    asm volatile("s_wait_asynccnt 0" ::: "memory");
}

// ---------------------------------------------------------------- zero output
__global__ void zero_f32_kernel(float* __restrict__ p, int n) {
    int i = blockIdx.x * 256 + threadIdx.x;
    if (i < n) p[i] = 0.0f;
}

// ---------------------------------------------------------------- weight fp32 -> bf16 transpose (LDS tiled)
// in: [R][C] fp32 (per blockIdx.z slab), out: [C][R] bf16
__global__ void transpose_bf16_kernel(const float* __restrict__ in,
                                      unsigned short* __restrict__ out,
                                      int R, int C) {
    __shared__ float tile[32][33];
    const size_t slab = (size_t)blockIdx.z * R * C;
    in  += slab;
    out += slab;
    const int rbase = blockIdx.y * 32;
    const int cbase = blockIdx.x * 32;
    const int tid = threadIdx.x;
    const int r  = tid >> 3;            // 0..31
    const int c4 = (tid & 7) * 4;       // 0..28
    float4 v = *(const float4*)(in + (size_t)(rbase + r) * C + cbase + c4);
    tile[r][c4 + 0] = v.x; tile[r][c4 + 1] = v.y;
    tile[r][c4 + 2] = v.z; tile[r][c4 + 3] = v.w;
    __syncthreads();
    const int rr = tid >> 3;            // out row (= original col) 0..31
    const int x4 = (tid & 7) * 4;       // out col group (= original row)
    ushort4 o;
    o.x = f2bf(tile[x4 + 0][rr]);
    o.y = f2bf(tile[x4 + 1][rr]);
    o.z = f2bf(tile[x4 + 2][rr]);
    o.w = f2bf(tile[x4 + 3][rr]);
    *(ushort4*)(out + (size_t)(cbase + rr) * R + rbase + x4) = o;
}

// ---------------------------------------------------------------- gating
__global__ void gate_kernel(const float* __restrict__ x,
                            const float* __restrict__ gw,
                            float* __restrict__ probs,
                            int*   __restrict__ topi,
                            float* __restrict__ topg) {
    __shared__ float red[N_EXP][256];
    const int t   = blockIdx.x;
    const int tid = threadIdx.x;
    const float* xr = x + (size_t)t * H_DIM;
    const int c0 = tid * 4;                       // H = 256 * 4
    float4 xv = *(const float4*)(xr + c0);
    float acc[N_EXP];
#pragma unroll
    for (int e = 0; e < N_EXP; ++e) {
        float4 g = *(const float4*)(gw + (size_t)e * H_DIM + c0);
        acc[e] = xv.x * g.x + xv.y * g.y + xv.z * g.z + xv.w * g.w;
    }
#pragma unroll
    for (int e = 0; e < N_EXP; ++e) red[e][tid] = acc[e];
    __syncthreads();
    for (int s = 128; s > 0; s >>= 1) {
        if (tid < s) {
#pragma unroll
            for (int e = 0; e < N_EXP; ++e) red[e][tid] += red[e][tid + s];
        }
        __syncthreads();
    }
    if (tid == 0) {
        float lg[N_EXP];
#pragma unroll
        for (int e = 0; e < N_EXP; ++e) lg[e] = red[e][0];
        float m = lg[0];
#pragma unroll
        for (int e = 1; e < N_EXP; ++e) m = fmaxf(m, lg[e]);
        float s = 0.f, p[N_EXP];
#pragma unroll
        for (int e = 0; e < N_EXP; ++e) { p[e] = expf(lg[e] - m); s += p[e]; }
        float inv = 1.0f / s;
#pragma unroll
        for (int e = 0; e < N_EXP; ++e) probs[(size_t)t * N_EXP + e] = p[e] * inv;
        int i0 = 0;
#pragma unroll
        for (int e = 1; e < N_EXP; ++e) if (lg[e] > lg[i0]) i0 = e;
        int i1 = (i0 == 0) ? 1 : 0;
#pragma unroll
        for (int e = 0; e < N_EXP; ++e) if (e != i0 && lg[e] > lg[i1]) i1 = e;
        float ee = expf(lg[i1] - lg[i0]);
        float g0 = 1.0f / (1.0f + ee);
        topi[t * 2 + 0] = i0;  topi[t * 2 + 1] = i1;
        topg[t * 2 + 0] = g0;  topg[t * 2 + 1] = 1.0f - g0;
    }
}

// ---------------------------------------------------------------- aux loss (deterministic)
__global__ void aux_kernel(const float* __restrict__ probs, float* __restrict__ out_aux) {
    __shared__ float sp[256][N_EXP];
    __shared__ float si[256][N_EXP];
    const int tid = threadIdx.x;
    float lp[N_EXP], li[N_EXP];
#pragma unroll
    for (int e = 0; e < N_EXP; ++e) { lp[e] = 0.f; li[e] = 0.f; }
    for (int t = tid; t < T_TOK; t += 256) {
#pragma unroll
        for (int e = 0; e < N_EXP; ++e) {
            float p = probs[(size_t)t * N_EXP + e];
            lp[e] += p;
            li[e] += (p > 0.0f) ? 1.0f : 0.0f;
        }
    }
#pragma unroll
    for (int e = 0; e < N_EXP; ++e) { sp[tid][e] = lp[e]; si[tid][e] = li[e]; }
    __syncthreads();
    for (int s = 128; s > 0; s >>= 1) {
        if (tid < s) {
#pragma unroll
            for (int e = 0; e < N_EXP; ++e) {
                sp[tid][e] += sp[tid + s][e];
                si[tid][e] += si[tid + s][e];
            }
        }
        __syncthreads();
    }
    if (tid == 0) {
        const float invT = 1.0f / (float)T_TOK;
        float aux = 0.f;
#pragma unroll
        for (int e = 0; e < N_EXP; ++e) aux += (sp[0][e] * invT) * (si[0][e] * invT);
        out_aux[0] = aux * (float)N_EXP;
    }
}

// ---------------------------------------------------------------- routing rank scan (deterministic cumsum order)
__global__ void rank_kernel(const int* __restrict__ topi,
                            int* __restrict__ slot,
                            int* __restrict__ keep) {
    __shared__ int cnt[N_EXP];
    __shared__ int wcnt[8][N_EXP];
    __shared__ int wbase[8][N_EXP];
    const int tid  = threadIdx.x;
    const int lane = tid & 31;
    const int wv   = tid >> 5;
    const unsigned lmask = (lane == 0) ? 0u : ((1u << lane) - 1u);
    for (int k = 0; k < TOPK; ++k) {
        if (tid < N_EXP) cnt[tid] = 0;
        __syncthreads();
        for (int c = 0; c < T_TOK / 256; ++c) {
            const int t = c * 256 + tid;
            const int e = topi[t * 2 + k];
            int rinw = 0;
#pragma unroll
            for (int e2 = 0; e2 < N_EXP; ++e2) {
                unsigned bm = (unsigned)__ballot(e == e2);     // wave32: low 32 bits
                if (lane == 0) wcnt[wv][e2] = __popc(bm);
                if (e == e2)   rinw = __popc(bm & lmask);
            }
            __syncthreads();
            if (tid < N_EXP) {
                const int e2 = tid;
                int run = cnt[e2];
#pragma unroll
                for (int w = 0; w < 8; ++w) { wbase[w][e2] = run; run += wcnt[w][e2]; }
                cnt[e2] = run;
            }
            __syncthreads();
            const int r  = wbase[wv][e] + rinw;
            const int kp = (r < CAP) ? 1 : 0;
            slot[t * 2 + k] = e * CAP + (kp ? r : (CAP - 1));
            keep[t * 2 + k] = kp;
            __syncthreads();
        }
        __syncthreads();
    }
}

// ---------------------------------------------------------------- scatter tokens -> bf16 capacity buffers
__global__ void scatter_kernel(const float* __restrict__ x,
                               const int* __restrict__ slot,
                               const int* __restrict__ keep,
                               unsigned short* __restrict__ xbuf) {
    const int t = blockIdx.x, k = blockIdx.y;
    if (!keep[t * 2 + k]) return;
    const int s = slot[t * 2 + k];
    const float* src = x + (size_t)t * H_DIM;
    unsigned short* dst = xbuf + ((size_t)k * N_EXP * CAP + (size_t)s) * H_DIM;
    const int c = threadIdx.x * 4;
    float4 v = *(const float4*)(src + c);
    ushort4 o;
    o.x = f2bf(v.x); o.y = f2bf(v.y); o.z = f2bf(v.z); o.w = f2bf(v.w);
    *(ushort4*)(dst + c) = o;
}

// ---------------------------------------------------------------- WMMA GEMM
// C[M x N] = act(A[M x K](bf16) * BT[N x K](bf16) + bias)
// 128x128 block tile, 8 wave32s as 4(M) x 2(N), wave tile 32x64 (2x4 WMMAs),
// BK = 64 (two 32-K substeps per staged tile -> 16 WMMAs per barrier).
// Double-buffered LDS; both operands staged with global_load_async_to_lds_b128.
#define BKK 64
#define AST 72                        // ushorts per row (144B: 16B-aligned, conflict-free)

template <bool GELU_BF16_OUT>
__global__ __launch_bounds__(256) void gemm_kernel(
        const unsigned short* __restrict__ A,    // [M][Kdim] bf16
        const unsigned short* __restrict__ BT,   // [Ntot][Kdim] bf16 (pre-transposed weights)
        const float* __restrict__ bias,          // [Ntot]
        void* __restrict__ C,                    // bf16[M x N] or f32[M x N]
        int Kdim, int Ntot) {
    __shared__ __align__(16) unsigned short As[2][128 * AST];
    __shared__ __align__(16) unsigned short Bt[2][128 * AST];

    const int tid  = threadIdx.x;
    const int lane = tid & 31;
    const int wv   = tid >> 5;
    const int wm   = wv & 3;                     // 4 waves along M
    const int wn   = wv >> 2;                    // 2 waves along N
    const int half = lane >> 4;
    const int l15  = lane & 15;
    const int tileM = blockIdx.y * 128;
    const int tileN = blockIdx.x * 128;

    auto stage = [&](int kb, int buf) {
        unsigned abase = (unsigned)(size_t)(&As[buf][0]);
        unsigned bbase = (unsigned)(size_t)(&Bt[buf][0]);
        // 128 rows x 64 bf16 = 1024 x 16B chunks per operand, 4 per thread
#pragma unroll
        for (int it = 0; it < 4; ++it) {
            const int idx = tid + it * 256;
            const int row = idx >> 3, q = idx & 7;
            async_copy_b128(abase + (unsigned)(row * AST + q * 8) * 2,
                            A + (size_t)(tileM + row) * Kdim + kb + q * 8);
            async_copy_b128(bbase + (unsigned)(row * AST + q * 8) * 2,
                            BT + (size_t)(tileN + row) * Kdim + kb + q * 8);
        }
    };

    v8f acc[2][4];
#pragma unroll
    for (int mt = 0; mt < 2; ++mt)
#pragma unroll
        for (int nt = 0; nt < 4; ++nt) acc[mt][nt] = {};

    stage(0, 0);
    wait_async0();
    __syncthreads();

    for (int kb = 0; kb < Kdim; kb += BKK) {
        const int cur = (kb / BKK) & 1;
        const bool more = (kb + BKK < Kdim);
        if (more) stage(kb + BKK, 1 - cur);      // overlap next tile with WMMA

#pragma unroll
        for (int ks = 0; ks < BKK; ks += 32) {
            // fragments: two contiguous b128 LDS loads each (ISA VGPR layouts)
            v16bf afrg[2], bfrg[4];
#pragma unroll
            for (int mt = 0; mt < 2; ++mt) {
                const unsigned short* ap =
                    &As[cur][(wm * 32 + mt * 16 + l15) * AST + ks + half * 8];
                Frag128 f;
                f.q[0] = *(const uint4*)ap;          // K = ks + 8*half + 0..7
                f.q[1] = *(const uint4*)(ap + 16);   // K = ks + 16 + 8*half + 0..7
                afrg[mt] = f.v;
            }
#pragma unroll
            for (int nt = 0; nt < 4; ++nt) {
                const unsigned short* bp =
                    &Bt[cur][(wn * 64 + nt * 16 + l15) * AST + ks + half * 8];
                Frag128 f;
                f.q[0] = *(const uint4*)bp;          // K = ks + 8*half + 0..7
                f.q[1] = *(const uint4*)(bp + 16);   // K = ks + 16 + 8*half + 0..7
                bfrg[nt] = f.v;
            }
#pragma unroll
            for (int mt = 0; mt < 2; ++mt)
#pragma unroll
                for (int nt = 0; nt < 4; ++nt)
                    acc[mt][nt] = __builtin_amdgcn_wmma_f32_16x16x32_bf16(
                        false, afrg[mt], false, bfrg[nt], (short)0, acc[mt][nt], false, false);
        }

        if (more) {
            wait_async0();              // next tiles resident
            __syncthreads();            // all waves done with cur buffer
        }
    }

    // epilogue: C/D layout -> M = r + 8*half, N = l15.  Bias hoisted to regs.
    float bv[4];
#pragma unroll
    for (int nt = 0; nt < 4; ++nt)
        bv[nt] = bias[tileN + wn * 64 + nt * 16 + l15];

#pragma unroll
    for (int mt = 0; mt < 2; ++mt) {
#pragma unroll
        for (int nt = 0; nt < 4; ++nt) {
            const int gc = tileN + wn * 64 + nt * 16 + l15;
#pragma unroll
            for (int r = 0; r < 8; ++r) {
                const int gr = tileM + wm * 32 + mt * 16 + r + half * 8;
                float v = acc[mt][nt][r] + bv[nt];
                if (GELU_BF16_OUT) {
                    ((unsigned short*)C)[(size_t)gr * Ntot + gc] = f2bf(gelu_f(v));
                } else {
                    ((float*)C)[(size_t)gr * Ntot + gc] = v;
                }
            }
        }
    }
}

// ---------------------------------------------------------------- gather expert outputs back to tokens
__global__ void gather_kernel(const float* __restrict__ eo,
                              const int* __restrict__ topi,
                              const int* __restrict__ slot,
                              const int* __restrict__ keep,
                              const float* __restrict__ topg,
                              float* __restrict__ out, int k, int e) {
    const int t = blockIdx.x;
    if (topi[t * 2 + k] != e || !keep[t * 2 + k]) return;
    const int row = slot[t * 2 + k] - e * CAP;
    const float g = topg[t * 2 + k];
    const int c = threadIdx.x * 4;
    float4 v = *(const float4*)(eo + (size_t)row * H_DIM + c);
    float* o = out + (size_t)t * H_DIM + c;
    float4 cur = *(float4*)o;
    cur.x += g * v.x; cur.y += g * v.y; cur.z += g * v.z; cur.w += g * v.w;
    *(float4*)o = cur;
}

// ---------------------------------------------------------------- launch
extern "C" void kernel_launch(void* const* d_in, const int* in_sizes, int n_in,
                              void* d_out, int out_size, void* d_ws, size_t ws_size,
                              hipStream_t stream) {
    (void)in_sizes; (void)n_in; (void)ws_size;
    const float* hs = (const float*)d_in[0];   // [B,S,H]
    const float* gw = (const float*)d_in[1];   // [E,H]
    const float* w1 = (const float*)d_in[2];   // [E,H,I]
    const float* b1 = (const float*)d_in[3];   // [E,I]
    const float* w2 = (const float*)d_in[4];   // [E,I,H]
    const float* b2 = (const float*)d_in[5];   // [E,H]
    float* out = (float*)d_out;

    char* w = (char*)d_ws;
    size_t off = 0;
    float* probs = (float*)(w + off); off += (size_t)T_TOK * N_EXP * 4;
    int*   topi  = (int*)(w + off);   off += (size_t)T_TOK * TOPK * 4;
    float* topg  = (float*)(w + off); off += (size_t)T_TOK * TOPK * 4;
    int*   slot  = (int*)(w + off);   off += (size_t)T_TOK * TOPK * 4;
    int*   keep  = (int*)(w + off);   off += (size_t)T_TOK * TOPK * 4;
    unsigned short* xbuf = (unsigned short*)(w + off); off += (size_t)TOPK * N_EXP * CAP * H_DIM * 2;
    unsigned short* hbuf = (unsigned short*)(w + off); off += (size_t)CAP * I_DIM * 2;
    float* eobuf = (float*)(w + off); off += (size_t)CAP * H_DIM * 4;
    unsigned short* w1t = (unsigned short*)(w + off); off += (size_t)N_EXP * H_DIM * I_DIM * 2;
    unsigned short* w2t = (unsigned short*)(w + off); off += (size_t)N_EXP * I_DIM * H_DIM * 2;

    // one-time per launch: weights fp32 -> bf16, transposed to N-major
    transpose_bf16_kernel<<<dim3(I_DIM / 32, H_DIM / 32, N_EXP), 256, 0, stream>>>(w1, w1t, H_DIM, I_DIM);
    transpose_bf16_kernel<<<dim3(H_DIM / 32, I_DIM / 32, N_EXP), 256, 0, stream>>>(w2, w2t, I_DIM, H_DIM);

    zero_f32_kernel<<<(out_size + 255) / 256, 256, 0, stream>>>(out, out_size);
    gate_kernel<<<T_TOK, 256, 0, stream>>>(hs, gw, probs, topi, topg);
    aux_kernel<<<1, 256, 0, stream>>>(probs, out + (size_t)T_TOK * H_DIM);
    rank_kernel<<<1, 256, 0, stream>>>(topi, slot, keep);
    scatter_kernel<<<dim3(T_TOK, TOPK), 256, 0, stream>>>(hs, slot, keep, xbuf);

    for (int k = 0; k < TOPK; ++k) {
        for (int e = 0; e < N_EXP; ++e) {
            const unsigned short* Xe = xbuf + ((size_t)k * N_EXP * CAP + (size_t)e * CAP) * H_DIM;
            gemm_kernel<true><<<dim3(I_DIM / 128, CAP / 128), 256, 0, stream>>>(
                Xe, w1t + (size_t)e * H_DIM * I_DIM, b1 + (size_t)e * I_DIM, hbuf, H_DIM, I_DIM);
            gemm_kernel<false><<<dim3(H_DIM / 128, CAP / 128), 256, 0, stream>>>(
                hbuf, w2t + (size_t)e * I_DIM * H_DIM, b2 + (size_t)e * H_DIM, eobuf, I_DIM, H_DIM);
            gather_kernel<<<T_TOK, 256, 0, stream>>>(eobuf, topi, slot, keep, topg, out, k, e);
        }
    }
}
